// VGG_wfc_34823594836437
// MI455X (gfx1250) — compile-verified, matrix operand order checked by reading, not run on hardware
//
#include <hip/hip_runtime.h>
#include <hip/hip_bf16.h>
#include <stdint.h>

typedef _Float16 half_t;
typedef __attribute__((ext_vector_type(16))) _Float16 v16h;
typedef __attribute__((ext_vector_type(8)))  float    v8f;

#define NLAYERS 16
#define BATCH 8
#define USE_ASYNC_B 1   // gfx1250 global_load_async_to_lds_b128 path for weight tiles

// ---------------------------------------------------------------------------
// Input packing: fp32 NCHW (8,3,256,256) -> f16 NHWC with C padded to 8
// ---------------------------------------------------------------------------
__global__ void pack_input_kernel(const float* __restrict__ x, half_t* __restrict__ out, int total) {
    int idx = blockIdx.x * blockDim.x + threadIdx.x;
    if (idx >= total) return;
    int c = idx & 7;
    int xx = (idx >> 3) & 255;
    int yy = (idx >> 11) & 255;
    int b = idx >> 19;
    float v = 0.0f;
    if (c < 3) v = x[(((size_t)b * 3 + c) * 256 + yy) * 256 + xx];
    out[idx] = (half_t)v;
}

// ---------------------------------------------------------------------------
// Weight packing directly into the per-K-step LDS tile image:
//   layer layout: [nt][kt] tiles, each tile = 128 rows (n) x 32 halves (k)
//   row stride 64B; dword kw stored at kw ^ (n & 15)  (bank-conflict-free reads)
//   k = (r*3+s)*Cpad + cin ;  n >= Cout rows are zero.
// One thread per packed dword (2 halves).
// ---------------------------------------------------------------------------
__device__ inline float wval(const float* __restrict__ w, int k, int o,
                             int Cin, int Cpad, int Cout) {
    if (o >= Cout) return 0.0f;
    int rs = k / Cpad;
    int cin = k - rs * Cpad;
    if (rs >= 9 || cin >= Cin) return 0.0f;
    int r = rs / 3, s = rs - 3 * (rs / 3);
    return w[(((size_t)o * Cin + cin) * 3 + r) * 3 + s];
}

__global__ void pack_weights_kernel(const float* __restrict__ w, half_t* __restrict__ wp,
                                    int Cin, int Cpad, int Cout, int ktiles, int total_dw) {
    int idx = blockIdx.x * blockDim.x + threadIdx.x;
    if (idx >= total_dw) return;
    int kw = idx & 15;            // dword within row (2 halves)
    int n  = (idx >> 4) & 127;    // row within tile
    int tile = idx >> 11;         // nt*ktiles + kt
    int kt = tile % ktiles;
    int nt = tile / ktiles;
    int o = nt * 128 + n;
    int k = kt * 32 + kw * 2;
    half_t two[2];
    two[0] = (half_t)wval(w, k,     o, Cin, Cpad, Cout);
    two[1] = (half_t)wval(w, k + 1, o, Cin, Cpad, Cout);
    int dst = tile * 2048 + n * 16 + (kw ^ (n & 15));   // dwords
    reinterpret_cast<unsigned int*>(wp)[dst] = *reinterpret_cast<unsigned int*>(two);
}

// ---------------------------------------------------------------------------
// Fold conv bias + BN: out = relu(conv*scale + shift)
// ---------------------------------------------------------------------------
__global__ void prep_scale_shift_kernel(const float* __restrict__ cb, const float* __restrict__ g,
                                        const float* __restrict__ bb, const float* __restrict__ m,
                                        const float* __restrict__ v, float* __restrict__ scale,
                                        float* __restrict__ shift, int C) {
    int c = blockIdx.x * blockDim.x + threadIdx.x;
    if (c >= C) return;
    float s = g[c] * rsqrtf(v[c] + 1e-5f);
    scale[c] = s;
    shift[c] = bb[c] + (cb[c] - m[c]) * s;
}

// ---------------------------------------------------------------------------
// Implicit-GEMM 3x3 SAME conv + BN + ReLU, V_WMMA_F32_16X16X32_F16.
// Macro tile 128(M) x 128(N) x 32(K); 8 waves = 2(M) x 4(N);
// per wave: 4 A-frags x 2 B-frags -> 8 WMMA / K-step; LDS double-buffered;
// B tile arrives via global_load_async_to_lds_b128 (ASYNCcnt).
// ---------------------------------------------------------------------------
#define BM 128
#define BN 128
#define BK 32
#define LDA_S (BK + 8)            // halves
#define BTILE_HALF 4096           // 128 * 32 halves = 8192 B

__global__ __launch_bounds__(256, 2)
void conv_wmma_kernel(const half_t* __restrict__ in, const half_t* __restrict__ wp,
                      const float* __restrict__ scale, const float* __restrict__ shift,
                      half_t* __restrict__ out, int Hdim, int CinP, int Cout,
                      int Kpad, int ktiles) {
    __shared__ half_t lA[2][BM * LDA_S];
    __shared__ half_t lB[2][BTILE_HALF];

    const int W  = Hdim;
    const int HW = Hdim * Hdim;
    const int b  = blockIdx.z;
    const int m0 = blockIdx.x * BM;
    const int n0tile = blockIdx.y;            // nt
    const int n0 = n0tile * BN;
    const int t  = threadIdx.x;
    const int lane   = t & 31;
    const int wave   = t >> 5;
    const int wmW    = wave & 1;              // 0..1 : M half (64 rows)
    const int wnW    = wave >> 1;             // 0..3 : N quarter (32 cols)
    const int laneHi = (lane >> 4) & 1;
    const int lmod   = lane & 15;

    v8f acc[8];
#pragma unroll
    for (int j = 0; j < 8; ++j) acc[j] = (v8f){0,0,0,0,0,0,0,0};

    // ---- staging helpers -------------------------------------------------
    // A: 128 rows x 32 halves, 512 16B-gathers, 2 per thread.
    // B: linear 8192B async copy from prepacked tile, 2 x B128 per thread.
    const size_t wtile_base = (size_t)n0tile * ktiles;          // tile index base

    auto stageA = [&](int kt, int buf) {
#pragma unroll
        for (int l = 0; l < 2; ++l) {
            int li = t + 256 * l;             // 0..511
            int am = li >> 2;                 // 0..127
            int kq = (li & 3) * 8;            // 0,8,16,24
            int gm = m0 + am;
            int ay = gm / W;
            int ax = gm - ay * W;
            int k  = kt * BK + kq;
            int rs = k / CinP;
            int cin = k - rs * CinP;
            uint4 val = make_uint4(0u, 0u, 0u, 0u);
            if (rs < 9 && gm < HW) {
                int r = rs / 3, s = rs - 3 * (rs / 3);
                int gy = ay + r - 1, gx = ax + s - 1;
                if ((unsigned)gy < (unsigned)Hdim && (unsigned)gx < (unsigned)W) {
                    val = *reinterpret_cast<const uint4*>(
                        in + ((((size_t)b * Hdim + gy) * W + gx) * CinP + cin));
                }
            }
            *reinterpret_cast<uint4*>(&lA[buf][am * LDA_S + kq]) = val;
        }
    };

    auto stageB = [&](int kt, int buf) {
        const half_t* src = wp + (wtile_base + kt) * (size_t)BTILE_HALF + (size_t)t * 8;
#if USE_ASYNC_B
        unsigned dst0 = (unsigned)(uintptr_t)(void*)(&lB[buf][t * 8]);
        unsigned dst1 = (unsigned)(uintptr_t)(void*)(&lB[buf][t * 8 + 2048]);
        uint64_t s0 = (uint64_t)(uintptr_t)src;
        uint64_t s1 = (uint64_t)(uintptr_t)(src + 2048);
        asm volatile("global_load_async_to_lds_b128 %0, %1, off"
                     :: "v"(dst0), "v"(s0) : "memory");
        asm volatile("global_load_async_to_lds_b128 %0, %1, off"
                     :: "v"(dst1), "v"(s1) : "memory");
#else
        uint4 v0 = *reinterpret_cast<const uint4*>(src);
        uint4 v1 = *reinterpret_cast<const uint4*>(src + 2048);
        *reinterpret_cast<uint4*>(&lB[buf][t * 8])        = v0;
        *reinterpret_cast<uint4*>(&lB[buf][t * 8 + 2048]) = v1;
#endif
    };

    auto waitB = [&]() {
#if USE_ASYNC_B
        asm volatile("s_wait_asynccnt 0" ::: "memory");
#endif
    };

    // ---- pipelined main loop --------------------------------------------
    stageA(0, 0);
    stageB(0, 0);
    waitB();
    __syncthreads();

    const int p0 = laneHi ? 4 : 0;     // dword-pair bases of 16-bit frag layout
    const int p1 = laneHi ? 12 : 8;

    for (int kt = 0; kt < ktiles; ++kt) {
        int cb = kt & 1;
        if (kt + 1 < ktiles) {
            stageA(kt + 1, cb ^ 1);
            stageB(kt + 1, cb ^ 1);
        }

        // B fragments (2), swizzled dword reads: physical = p ^ (row & 15)
        v16h bfrag[2];
#pragma unroll
        for (int nj = 0; nj < 2; ++nj) {
            int row = wnW * 32 + nj * 16 + lmod;
            const unsigned int* brow =
                reinterpret_cast<const unsigned int*>(&lB[cb][row * BK]);
            unsigned int* bu = reinterpret_cast<unsigned int*>(&bfrag[nj]);
#pragma unroll
            for (int e = 0; e < 4; ++e) {
                bu[e]     = brow[(p0 + e) ^ lmod];
                bu[4 + e] = brow[(p1 + e) ^ lmod];
            }
        }
        // A fragments (4), each feeding 2 WMMAs
#pragma unroll
        for (int mi = 0; mi < 4; ++mi) {
            v16h afrag;
            const unsigned int* arow = reinterpret_cast<const unsigned int*>(
                &lA[cb][(wmW * 64 + mi * 16 + lmod) * LDA_S]);
            unsigned int* au = reinterpret_cast<unsigned int*>(&afrag);
#pragma unroll
            for (int e = 0; e < 4; ++e) { au[e] = arow[p0 + e]; au[4 + e] = arow[p1 + e]; }
            acc[mi * 2 + 0] = __builtin_amdgcn_wmma_f32_16x16x32_f16(
                false, afrag, false, bfrag[0], (short)0, acc[mi * 2 + 0], false, false);
            acc[mi * 2 + 1] = __builtin_amdgcn_wmma_f32_16x16x32_f16(
                false, afrag, false, bfrag[1], (short)0, acc[mi * 2 + 1], false, false);
        }
        waitB();
        __syncthreads();
    }

    // ---- fused BN+ReLU epilogue, f16 NHWC store --------------------------
#pragma unroll
    for (int nj = 0; nj < 2; ++nj) {
        int n = n0 + wnW * 32 + nj * 16 + lmod;
        if (n >= Cout) continue;
        float sc = scale[n], sh = shift[n];
#pragma unroll
        for (int mi = 0; mi < 4; ++mi) {
#pragma unroll
            for (int r = 0; r < 8; ++r) {
                int m = m0 + wmW * 64 + mi * 16 + r + (laneHi ? 8 : 0);
                if (m < HW) {
                    float v = acc[mi * 2 + nj][r] * sc + sh;
                    v = v > 0.0f ? v : 0.0f;
                    out[((size_t)b * HW + m) * Cout + n] = (half_t)v;
                }
            }
        }
    }
}

// ---------------------------------------------------------------------------
// 2x2 maxpool, NHWC f16
// ---------------------------------------------------------------------------
__global__ void maxpool_kernel(const half_t* __restrict__ in, half_t* __restrict__ out,
                               int Hin, int C, int total) {
    int idx = blockIdx.x * blockDim.x + threadIdx.x;
    if (idx >= total) return;
    int Ho = Hin >> 1;
    int c = idx % C;  int r = idx / C;
    int xo = r % Ho;  r /= Ho;
    int yo = r % Ho;  int b = r / Ho;
    const half_t* p = in + (((size_t)b * Hin + yo * 2) * Hin + xo * 2) * C + c;
    float v0 = (float)p[0];
    float v1 = (float)p[C];
    float v2 = (float)p[(size_t)Hin * C];
    float v3 = (float)p[(size_t)Hin * C + C];
    out[idx] = (half_t)fmaxf(fmaxf(v0, v1), fmaxf(v2, v3));
}

// ---------------------------------------------------------------------------
// FC: l_weight[b][j] = sum act[b][m][c] * fc_w[j][c*64+m] + fc_b[j]
// ---------------------------------------------------------------------------
__global__ __launch_bounds__(256)
void fc_kernel(const half_t* __restrict__ act, const float* __restrict__ fw,
               const float* __restrict__ fb, float* __restrict__ lw) {
    __shared__ float red[256];
    int b = blockIdx.x, j = blockIdx.y, t = threadIdx.x;
    float acc = 0.0f;
    for (int i = t; i < 32768; i += 256) {
        int m = i >> 9;
        int c = i & 511;
        float a = (float)act[((size_t)b * 64 + m) * 512 + c];
        acc += a * fw[(size_t)j * 32768 + c * 64 + m];
    }
    red[t] = acc;
    __syncthreads();
    for (int s = 128; s > 0; s >>= 1) { if (t < s) red[t] += red[t + s]; __syncthreads(); }
    if (t == 0) lw[b * 10 + j] = red[0] + fb[j];
}

// ---------------------------------------------------------------------------
// 256-point radix-2 DIT FFT in LDS (128 threads, preloaded bit-reversed)
// ---------------------------------------------------------------------------
__device__ inline void fft256_stages(float2* buf, int t, float sign) {
#pragma unroll
    for (int len = 2; len <= 256; len <<= 1) {
        int half = len >> 1;
        int blk = t / half;
        int pos = t - blk * half;
        int i = blk * len + pos;
        int j = i + half;
        float ang = sign * 6.28318530717958647692f * (float)pos / (float)len;
        float sn, cs;
        __sincosf(ang, &sn, &cs);
        float2 u = buf[i], w = buf[j];
        float2 wv = make_float2(w.x * cs - w.y * sn, w.x * sn + w.y * cs);
        buf[i] = make_float2(u.x + wv.x, u.y + wv.y);
        buf[j] = make_float2(u.x - wv.x, u.y - wv.y);
        __syncthreads();
    }
}

__global__ __launch_bounds__(128)
void fft_rows_fwd_kernel(const float* __restrict__ x, float2* __restrict__ work) {
    __shared__ float2 buf[256];
    int line = blockIdx.x;
    int t = threadIdx.x;
    const float* row = x + (size_t)line * 256;
    buf[__brev((unsigned)t) >> 24]         = make_float2(row[t], 0.0f);
    buf[__brev((unsigned)(t + 128)) >> 24] = make_float2(row[t + 128], 0.0f);
    __syncthreads();
    fft256_stages(buf, t, -1.0f);
    float2* orow = work + (size_t)line * 256;
    orow[t] = buf[t];
    orow[t + 128] = buf[t + 128];
}

__global__ __launch_bounds__(128)
void fft_cols_filter_kernel(float2* __restrict__ work, const float* __restrict__ lw) {
    __shared__ float2 buf[256];
    int col = blockIdx.x;
    int x = col & 255;
    int plane = col >> 8;
    int b = plane / 3;
    int t = threadIdx.x;
    float2* base = work + (size_t)plane * 65536 + x;

    buf[__brev((unsigned)t) >> 24]         = base[(size_t)t * 256];
    buf[__brev((unsigned)(t + 128)) >> 24] = base[(size_t)(t + 128) * 256];
    __syncthreads();
    fft256_stages(buf, t, -1.0f);

    int cv = (x < 128) ? x : x - 256;
    const float bw = 256.0f / 2.0f / 9.0f;
#pragma unroll
    for (int q = 0; q < 2; ++q) {
        int u = t + q * 128;
        int cu = (u < 128) ? u : u - 256;
        float r = sqrtf((float)(cu * cu + cv * cv));
        int idx = (int)floorf(r / bw);
        int band = 9;
        if (idx < 9 && r > (float)idx * bw && r < (float)(idx + 1) * bw) band = idx;
        float f = lw[b * 10 + band];
        buf[u].x *= f;
        buf[u].y *= f;
    }
    __syncthreads();
    float2 v0 = buf[t], v1 = buf[t + 128];
    __syncthreads();
    buf[__brev((unsigned)t) >> 24]         = v0;
    buf[__brev((unsigned)(t + 128)) >> 24] = v1;
    __syncthreads();
    fft256_stages(buf, t, +1.0f);

    const float invN = 1.0f / 256.0f;
    base[(size_t)t * 256]         = make_float2(buf[t].x * invN, buf[t].y * invN);
    base[(size_t)(t + 128) * 256] = make_float2(buf[t + 128].x * invN, buf[t + 128].y * invN);
}

__global__ __launch_bounds__(128)
void fft_rows_inv_abs_kernel(const float2* __restrict__ work, float* __restrict__ out) {
    __shared__ float2 buf[256];
    int line = blockIdx.x;
    int t = threadIdx.x;
    const float2* row = work + (size_t)line * 256;
    buf[__brev((unsigned)t) >> 24]         = row[t];
    buf[__brev((unsigned)(t + 128)) >> 24] = row[t + 128];
    __syncthreads();
    fft256_stages(buf, t, +1.0f);
    const float invN = 1.0f / 256.0f;
    float* orow = out + (size_t)line * 256;
    float2 a = buf[t];
    float2 c = buf[t + 128];
    orow[t]       = sqrtf(a.x * a.x + a.y * a.y) * invN;
    orow[t + 128] = sqrtf(c.x * c.x + c.y * c.y) * invN;
}

// ---------------------------------------------------------------------------
// Host orchestration
// ---------------------------------------------------------------------------
extern "C" void kernel_launch(void* const* d_in, const int* in_sizes, int n_in,
                              void* d_out, int out_size, void* d_ws, size_t ws_size,
                              hipStream_t stream) {
    (void)in_sizes; (void)n_in; (void)out_size; (void)ws_size;

    static const int COUT[NLAYERS] = {64,64,128,128,256,256,256,256,512,512,512,512,512,512,512,512};
    static const int CIN [NLAYERS] = {3,64,64,128,128,256,256,256,256,512,512,512,512,512,512,512};
    static const int HDIM[NLAYERS] = {256,256,128,128,64,64,64,64,32,32,32,32,16,16,16,16};
    static const bool POOL[NLAYERS] = {false,true,false,true,false,false,false,true,
                                       false,false,false,true,false,false,false,true};

    const float* x = (const float*)d_in[0];
    const float* conv_w[NLAYERS]; const float* conv_b[NLAYERS];
    const float* bn_g[NLAYERS];   const float* bn_b[NLAYERS];
    const float* bn_m[NLAYERS];   const float* bn_v[NLAYERS];
    for (int i = 0; i < NLAYERS; ++i) {
        conv_w[i] = (const float*)d_in[1 + i];
        conv_b[i] = (const float*)d_in[17 + i];
        bn_g[i]   = (const float*)d_in[33 + i];
        bn_b[i]   = (const float*)d_in[49 + i];
        bn_m[i]   = (const float*)d_in[65 + i];
        bn_v[i]   = (const float*)d_in[81 + i];
    }
    const float* fc_w = (const float*)d_in[97];
    const float* fc_b = (const float*)d_in[98];
    float* out = (float*)d_out;

    int CPAD[NLAYERS], KPAD[NLAYERS], KT[NLAYERS], NT[NLAYERS];
    size_t woff[NLAYERS], soff[NLAYERS];
    size_t wtot = 0, stot = 0;
    for (int i = 0; i < NLAYERS; ++i) {
        CPAD[i] = (i == 0) ? 8 : CIN[i];
        KPAD[i] = (i == 0) ? 96 : 9 * CIN[i];
        KT[i]   = KPAD[i] / 32;
        NT[i]   = (COUT[i] + 127) / 128;
        woff[i] = wtot;  wtot += (size_t)NT[i] * KT[i] * BTILE_HALF;   // halves
        soff[i] = stot;  stot += (size_t)COUT[i];
    }

    char* base = (char*)d_ws;
    size_t off = 0;
    auto carve = [&](size_t bytes) -> void* {
        void* p = base + off;
        off += (bytes + 255) & ~(size_t)255;
        return p;
    };
    half_t* bufA   = (half_t*)carve((size_t)BATCH * 256 * 256 * 64 * sizeof(half_t));
    half_t* bufB   = (half_t*)carve((size_t)BATCH * 256 * 256 * 64 * sizeof(half_t));
    half_t* wpack  = (half_t*)carve(wtot * sizeof(half_t));
    float*  scaleB = (float*)carve(stot * sizeof(float));
    float*  shiftB = (float*)carve(stot * sizeof(float));
    float*  lw     = (float*)carve(80 * sizeof(float));
    float2* fwork  = (float2*)carve((size_t)BATCH * 3 * 256 * 256 * sizeof(float2));

    // ---- prep ----
    {
        int total = BATCH * 256 * 256 * 8;
        pack_input_kernel<<<(total + 255) / 256, 256, 0, stream>>>(x, bufA, total);
    }
    for (int li = 0; li < NLAYERS; ++li) {
        int total_dw = NT[li] * KT[li] * 2048;   // dwords in packed image
        pack_weights_kernel<<<(total_dw + 255) / 256, 256, 0, stream>>>(
            conv_w[li], wpack + woff[li], CIN[li], CPAD[li], COUT[li], KT[li], total_dw);
        prep_scale_shift_kernel<<<(COUT[li] + 255) / 256, 256, 0, stream>>>(
            conv_b[li], bn_g[li], bn_b[li], bn_m[li], bn_v[li],
            scaleB + soff[li], shiftB + soff[li], COUT[li]);
    }

    // ---- VGG forward ----
    half_t* cur = bufA;
    half_t* nxt = bufB;
    for (int li = 0; li < NLAYERS; ++li) {
        int Hd = HDIM[li];
        int HW = Hd * Hd;
        dim3 grid((HW + BM - 1) / BM, NT[li], BATCH);
        conv_wmma_kernel<<<grid, 256, 0, stream>>>(
            cur, wpack + woff[li], scaleB + soff[li], shiftB + soff[li],
            nxt, Hd, CPAD[li], COUT[li], KPAD[li], KT[li]);
        { half_t* tmp = cur; cur = nxt; nxt = tmp; }
        if (POOL[li]) {
            int Ho = Hd >> 1;
            int total = BATCH * Ho * Ho * COUT[li];
            maxpool_kernel<<<(total + 255) / 256, 256, 0, stream>>>(cur, nxt, Hd, COUT[li], total);
            { half_t* tmp = cur; cur = nxt; nxt = tmp; }
        }
    }

    // ---- FC -> l_weight ----
    {
        dim3 grid(BATCH, 10);
        fc_kernel<<<grid, 256, 0, stream>>>(cur, fc_w, fc_b, lw);
    }

    // ---- spectral band filter ----
    {
        int lines = BATCH * 3 * 256;
        fft_rows_fwd_kernel<<<lines, 128, 0, stream>>>(x, fwork);
        fft_cols_filter_kernel<<<lines, 128, 0, stream>>>(fwork, lw);
        fft_rows_inv_abs_kernel<<<lines, 128, 0, stream>>>(fwork, out);
    }
}
